// HysteresisThresholding_3066606649584
// MI455X (gfx1250) — matrix operand order repaired and checked by reading, not compiled, not run on hardware
//
#include <hip/hip_runtime.h>

// Hysteresis thresholding for 2048x2048 f32, CDNA5/gfx1250.
// Flood-fill dilation implemented as band-matrix multiplies on the
// f16 WMMA path (V_WMMA_F32_16X16X32_F16), masks resident in L2.

#define H_IMG 2048
#define W_IMG 2048
#define NPIX  (H_IMG * W_IMG)
#define LOW_T  0.3f
#define HIGH_T 0.7f
#define ITERS  12          // even -> final active mask lands in buffer A

typedef __attribute__((ext_vector_type(16))) _Float16 v16h;
typedef __attribute__((ext_vector_type(8)))  float    v8f;

// ---------------------------------------------------------------------------
// Kernel 1: thresholds + seed mask.  out[0..N) = low, out[N..2N) = high,
// activeA = f16 0/1 seed mask (high > 0).
// ---------------------------------------------------------------------------
__global__ __launch_bounds__(256)
void hyst_threshold_init(const float* __restrict__ thin,
                         float* __restrict__ out,
                         _Float16* __restrict__ activeA) {
    int i = blockIdx.x * blockDim.x + threadIdx.x;
    if (i >= NPIX) return;
    float t  = thin[i];
    float lt = (t < LOW_T)  ? 0.0f : t;
    float hi = (t < HIGH_T) ? 0.0f : t;
    out[i]         = lt;
    out[NPIX + i]  = hi;
    activeA[i] = (hi > 0.0f) ? (_Float16)1.0f : (_Float16)0.0f;
}

// ---------------------------------------------------------------------------
// Kernel 2: one Jacobi sweep of the hysteresis flood fill.
// Separable binary dilation at gaps 1 and 2 via WMMA band-matrix multiplies.
// Block = 256 threads (8 waves), covers a 16-row x 128-col output strip.
// ---------------------------------------------------------------------------
__global__ __launch_bounds__(256)
void hyst_dilate_wmma(const _Float16* __restrict__ src,
                      _Float16* __restrict__ dst,
                      const float* __restrict__ thin) {
    // LDS: active neighborhood rows r0-8..r0+23 (32), cols c0-16..c0+143 (160)
    __shared__ _Float16 sAct[32 * 164];
    __shared__ _Float16 sV1[16 * 164];   // vertically dilated, gap 1 (0/1)
    __shared__ _Float16 sV2[16 * 164];   // vertically dilated, gap 2 (0/1)

    const int tid  = threadIdx.x;
    const int w    = tid >> 5;           // wave id 0..7 (wave32)
    const int lane = tid & 31;
    const int c0   = blockIdx.x * 128;   // first output column of this block
    const int r0   = blockIdx.y * 16;    // first output row of this block

    // Cooperative zero-padded load of the active mask neighborhood.
    for (int e = tid; e < 32 * 160; e += 256) {
        int row = e / 160, col = e % 160;
        int gr = r0 - 8 + row;
        int gc = c0 - 16 + col;
        _Float16 v = (_Float16)0.0f;
        if (gr >= 0 && gr < H_IMG && gc >= 0 && gc < W_IMG)
            v = src[gr * W_IMG + gc];
        sAct[row * 164 + col] = v;
    }
    __syncthreads();

    const int  M      = lane & 15;       // matrix row (A/D) or column (B/D)
    const bool hiHalf = (lane >= 16);

    // Constant A fragments: vertical band matrices Ay_g (16x32 f16).
    // A layout: lanes 0-15 hold K in {0..7,16..23}; lanes 16-31 K in {8..15,24..31}.
    v16h a1, a2;
#pragma unroll
    for (int e = 0; e < 16; ++e) {
        int K = hiHalf ? ((e < 8) ? e + 8 : e + 16)
                       : ((e < 8) ? e     : e + 8);
        a1[e] = (K == M + 7 || K == M + 8 || K == M + 9)  ? (_Float16)1.0f : (_Float16)0.0f;
        a2[e] = (K == M + 6 || K == M + 8 || K == M + 10) ? (_Float16)1.0f : (_Float16)0.0f;
    }

    const v8f zc = {0.f, 0.f, 0.f, 0.f, 0.f, 0.f, 0.f, 0.f};

    // Stage 1: vertical dilation, 10 column tiles by 8 waves (waves 0,1 double).
    for (int pass = 0; pass < 2; ++pass) {
        if (pass == 1 && w >= 2) break;                 // wave-uniform branch
        const int tcol = (pass == 0) ? w : (w + 8);
        const int base = tcol * 16;                     // LDS col base of tile

        // B fragment = Act(32x16): lanes 0-15 hold K in {0..7,16..23}; lanes 16-31 K in {8..15,24..31}.
        v16h bAct;
#pragma unroll
        for (int e = 0; e < 16; ++e) {
            int K = hiHalf ? ((e < 8) ? e + 8 : e + 16) : ((e < 8) ? e : e + 8);
            bAct[e] = sAct[K * 164 + base + M];
        }
        v8f d1 = __builtin_amdgcn_wmma_f32_16x16x32_f16(
            false, a1, false, bAct, (short)0, zc, false, false);
        v8f d2 = __builtin_amdgcn_wmma_f32_16x16x32_f16(
            false, a2, false, bAct, (short)0, zc, false, false);

        // D layout: lanes 0-15 M=v, lanes 16-31 M=v+8; N = lane&15.
#pragma unroll
        for (int v = 0; v < 8; ++v) {
            int m = v + (hiHalf ? 8 : 0);
            sV1[m * 164 + base + M] = (d1[v] > 0.5f) ? (_Float16)1.0f : (_Float16)0.0f;
            sV2[m * 164 + base + M] = (d2[v] > 0.5f) ? (_Float16)1.0f : (_Float16)0.0f;
        }
    }
    __syncthreads();

    // Constant B fragments: horizontal band matrices Bx_g (32x16).
    // Bx_g[k][n] = 1 iff k in {n+8-g, n+8, n+8+g};  N == lane&15 == M here.
    v16h bx1, bx2;
#pragma unroll
    for (int e = 0; e < 16; ++e) {
        int K = hiHalf ? ((e < 8) ? e + 8 : e + 16) : ((e < 8) ? e : e + 8);
        bx1[e] = (K == M + 7 || K == M + 8 || K == M + 9)  ? (_Float16)1.0f : (_Float16)0.0f;
        bx2[e] = (K == M + 6 || K == M + 8 || K == M + 10) ? (_Float16)1.0f : (_Float16)0.0f;
    }

    // Stage 2: horizontal dilation for this wave's 16x16 output tile.
    const int base2 = 16 + w * 16;       // LDS col of output tile start
    v16h aV1, aV2;
#pragma unroll
    for (int e = 0; e < 16; ++e) {
        int K = hiHalf ? ((e < 8) ? e + 8 : e + 16)
                       : ((e < 8) ? e     : e + 8);
        int col = base2 - 8 + K;         // Vwide col k <-> LDS col base2-8+k
        aV1[e] = sV1[M * 164 + col];
        aV2[e] = sV2[M * 164 + col];
    }
    v8f e1 = __builtin_amdgcn_wmma_f32_16x16x32_f16(
        false, aV1, false, bx1, (short)0, zc, false, false);
    v8f e2 = __builtin_amdgcn_wmma_f32_16x16x32_f16(
        false, aV2, false, bx2, (short)0, zc, false, false);

    // Combine: new = active | (weak & ((c1 & box1) | (c2 & box2))).
#pragma unroll
    for (int v = 0; v < 8; ++v) {
        int gr  = r0 + v + (hiHalf ? 8 : 0);
        int gc  = c0 + w * 16 + M;
        int idx = gr * W_IMG + gc;
        float t  = thin[idx];
        bool wk  = (t >= LOW_T);                    // low_thresholded > 0
        bool act = ((float)src[idx] > 0.5f);
        bool b1  = (e1[v] > 0.5f) && (gc >= 1) && (gr >= 1)
                   && (gc + 1 < W_IMG - 1) && (gr + 1 < H_IMG - 1);
        bool b2  = (e2[v] > 0.5f) && (gc >= 2) && (gr >= 2)
                   && (gc + 2 < W_IMG - 1) && (gr + 2 < H_IMG - 1);
        bool na  = act || (wk && (b1 || b2));
        dst[idx] = na ? (_Float16)1.0f : (_Float16)0.0f;
    }
}

// ---------------------------------------------------------------------------
// Kernel 3: final compose.  out[2N..3N) = hi>0 ? hi : (active ? lt : 0)
// ---------------------------------------------------------------------------
__global__ __launch_bounds__(256)
void hyst_finalize(const float* __restrict__ thin,
                   const _Float16* __restrict__ active,
                   float* __restrict__ out) {
    int i = blockIdx.x * blockDim.x + threadIdx.x;
    if (i >= NPIX) return;
    float t  = thin[i];
    float lt = (t < LOW_T)  ? 0.0f : t;
    float hi = (t < HIGH_T) ? 0.0f : t;
    bool act = ((float)active[i] > 0.5f);
    out[2 * NPIX + i] = (hi > 0.0f) ? hi : (act ? lt : 0.0f);
}

// ---------------------------------------------------------------------------
extern "C" void kernel_launch(void* const* d_in, const int* in_sizes, int n_in,
                              void* d_out, int out_size, void* d_ws, size_t ws_size,
                              hipStream_t stream) {
    (void)in_sizes; (void)n_in; (void)out_size; (void)ws_size;
    const float* thin = (const float*)d_in[0];   // thin_edges (only input used)
    float* out = (float*)d_out;

    _Float16* bufA = (_Float16*)d_ws;            // 8 MB
    _Float16* bufB = bufA + NPIX;                // 8 MB

    const int ethreads = 256;
    const int eblocks  = NPIX / ethreads;

    hyst_threshold_init<<<eblocks, ethreads, 0, stream>>>(thin, out, bufA);

    dim3 dgrid(W_IMG / 128, H_IMG / 16);
    for (int k = 0; k < ITERS; ++k) {
        const _Float16* s = (k & 1) ? bufB : bufA;
        _Float16*       d = (k & 1) ? bufA : bufB;
        hyst_dilate_wmma<<<dgrid, 256, 0, stream>>>(s, d, thin);
    }

    // ITERS even -> converged mask is in bufA
    hyst_finalize<<<eblocks, ethreads, 0, stream>>>(thin, bufA, out);
}
